// ResidualVectorQuantizer_15358803050823
// MI455X (gfx1250) — matrix operand order repaired and checked by reading, not compiled, not run on hardware
//
#include <hip/hip_runtime.h>
#include <hip/hip_bf16.h>

// ---------------- problem constants ----------------
#define BB    8        // batch
#define DD    256      // feature dim
#define TT    8192     // time steps
#define NQ    8        // residual stages
#define BINS  1024     // codebook size
#define NTOK  (BB*TT)  // 65536 rows

// tiling
#define ROWS      128             // token rows per block
#define RSTRIDE   264             // bf16 row stride in LDS (16B aligned, bank-friendly)
#define NTILES    (BINS/16)       // 64 N-tiles of 16 bins
#define KCHUNKS   (DD/32)         // 8 K-steps of 32
#define TILE_EL   (KCHUNKS*32*16) // 4096 bf16 = 8KB per B tile
#define TILE_BY   (TILE_EL*2)     // 8192 bytes
#define SLOTS     4               // async ring depth (prefetch distance 3)

// d_out layout (flat, in return order): quantized | codes(int32 bits) | penalty
#define QUANT_ELEMS ((size_t)BB*DD*TT)        // 16,777,216
#define CODES_ELEMS ((size_t)NQ*BB*TT)        //    524,288

// d_ws layout: [0] loss accum (float) | +256B: e2[8*1024] f32 | +256B+32KB: swizzled bf16 B
#define WS_E2_OFF  256
#define WS_B_OFF   (256 + (size_t)NQ*BINS*4)

typedef __bf16 bf16;
typedef __attribute__((ext_vector_type(16))) bf16  v16bf;
typedef __attribute__((ext_vector_type(8)))  bf16  v8bf;
typedef __attribute__((ext_vector_type(8)))  float v8f;

// gfx1250 async global->LDS copy (ASYNCcnt-tracked), CDNA5 ISA 10.x / 15.18.3.
// INST_OFFSET is added to BOTH the LDS and global address, so one base pair + offset
// covers 32B per lane. LDS operand = low 32 bits of the flat LDS pointer.
__device__ __forceinline__ void async_copy_32B(unsigned lds_off, const char* gptr) {
    asm volatile("global_load_async_to_lds_b128 %0, %1, off"
                 :: "v"(lds_off), "v"(gptr) : "memory");
    asm volatile("global_load_async_to_lds_b128 %0, %1, off offset:16"
                 :: "v"(lds_off), "v"(gptr) : "memory");
}
template <int N>
__device__ __forceinline__ void wait_async_le() {
    asm volatile("s_wait_asynccnt %0" :: "n"(N) : "memory");
}

// ---------- prep 1: |e_k|^2 per (stage,bin) + zero loss accumulator ----------
__global__ void rvq_prep_e2(const float* __restrict__ embed,
                            float* __restrict__ ws_e2, float* __restrict__ ws_loss) {
    int gid = blockIdx.x * 256 + threadIdx.x;      // 0 .. NQ*BINS-1
    if (gid == 0) ws_loss[0] = 0.0f;
    const float* ep = embed + (size_t)gid * DD;
    float acc = 0.0f;
    #pragma unroll 4
    for (int d = 0; d < DD; ++d) { float v = ep[d]; acc += v * v; }
    ws_e2[gid] = acc;
}

// ---------- prep 2: swizzle codebook fp32 -> bf16 in B-fragment lane order ----------
// flat bits: [s:3][nt:6][kc:3][lane:5][e:4]; lane&15 = column(bin), lane>>4 picks K half.
__global__ void rvq_prep_swz(const float* __restrict__ embed, bf16* __restrict__ ws_b) {
    int gid = blockIdx.x * 256 + threadIdx.x;      // 0 .. NQ*BINS*DD-1
    int e   =  gid        & 15;
    int l   = (gid >> 4)  & 31;
    int kc  = (gid >> 9)  & 7;
    int nt  = (gid >> 12) & 63;
    int s   =  gid >> 18;
    int bin = nt * 16 + (l & 15);
    int k   = kc * 32 + (l >> 4) * 16 + e;
    ws_b[gid] = (bf16)embed[((size_t)s * BINS + bin) * DD + k];
}

// ---------- main: 8-stage RVQ with residual resident in LDS ----------
__global__ void __launch_bounds__(256)
rvq_main(const float* __restrict__ x, const float* __restrict__ embed,
         const bf16* __restrict__ ws_b, const float* __restrict__ ws_e2,
         float* __restrict__ ws_loss,
         float* __restrict__ q_out, int* __restrict__ codes_out) {
    __shared__ __align__(64) bf16  lds_res[ROWS * RSTRIDE];   // resident residual (bf16)
    __shared__ __align__(64) bf16  btile[SLOTS][TILE_EL];     // async B ring (4x8KB)
    __shared__ __align__(16) float lds_e2[BINS];              // |e|^2 for current stage
    __shared__ int lds_idx[ROWS];

    const int tid  = threadIdx.x;            // 256 threads = 8 waves
    const int lane = tid & 31;
    const int wv   = tid >> 5;               // wave id 0..7
    const int r0   = wv * 16;                // 16-row strip per wave
    const int bb   = blockIdx.x >> 6;        // batch index
    const int t0   = (blockIdx.x & 63) << 7; // 128-token chunk

    // ---- load x tile -> LDS residual (bf16), coalesced over t ----
    for (int i = tid; i < ROWS * DD; i += 256) {
        int row = i & (ROWS - 1);
        int d   = i >> 7;
        float v = x[((size_t)bb * DD + d) * TT + t0 + row];
        lds_res[row * RSTRIDE + d] = (bf16)v;
    }
    __syncthreads();

    const int m    = lane & 15;
    const int half = lane >> 4;
    float lossAcc = 0.0f;

    // per-thread async slot base: 32B at element tid*16 of slot 0
    const unsigned my_lds = (unsigned)(uintptr_t)&btile[0][tid * 16];
    const bf16*    bl0    = &btile[0][lane * 16];

    for (int s = 0; s < NQ; ++s) {
        // ---- build resident A fragments (16x256 bf16 strip -> 8 x v16bf) ----
        v16bf afrag[KCHUNKS];
        const bf16* rbase = &lds_res[(r0 + m) * RSTRIDE];
        #pragma unroll
        for (int kc = 0; kc < KCHUNKS; ++kc) {
            union { v16bf v; v8bf h[2]; } u;
            u.h[0] = *(const v8bf*)(rbase + kc * 32 + half * 16);
            u.h[1] = *(const v8bf*)(rbase + kc * 32 + half * 16 + 8);
            afrag[kc] = u.v;
        }

        float best[8]; int bidx[8];
        #pragma unroll
        for (int v = 0; v < 8; ++v) { best[v] = -3.4e38f; bidx[v] = 0; }

        const char* bstage = (const char*)(ws_b + (size_t)s * NTILES * TILE_EL)
                             + (size_t)tid * 32;

        // ---- stage prologue: drain ring, stage |e|^2, prime 3 tiles ----
        wait_async_le<0>();                       // previous stage's ring fully landed
        for (int i = tid; i < BINS; i += 256)     // |e|^2 slice -> LDS
            lds_e2[i] = ws_e2[s * BINS + i];
        #pragma unroll
        for (int p = 0; p < SLOTS - 1; ++p)
            async_copy_32B(my_lds + p * TILE_BY, bstage + (size_t)p * TILE_BY);
        wait_async_le<4>();                       // oldest (tile 0) landed
        __syncthreads();                          // covers lds_e2 + tile 0 visibility

        // ---- sweep all 64 N-tiles; 4-slot async ring, distance-3 prefetch ----
        for (int nt = 0; nt < NTILES; ++nt) {
            // prefetch tile nt+3 into slot (nt+3)&3 (last read at iteration nt-1,
            // protected by that iteration's end barrier). Tail re-fetches tile 63
            // so the in-flight count stays uniform.
            int pf = nt + SLOTS - 1; if (pf > NTILES - 1) pf = NTILES - 1;
            async_copy_32B(my_lds + ((nt + SLOTS - 1) & (SLOTS - 1)) * TILE_BY,
                           bstage + (size_t)pf * TILE_BY);

            // compute from slot nt&3, 2-deep register pipeline over kc
            const bf16* bl = bl0 + (nt & (SLOTS - 1)) * TILE_EL;
            v16bf bcur = *(const v16bf*)bl;
            v8f c = {};
            #pragma unroll
            for (int kc = 0; kc < KCHUNKS; ++kc) {
                v16bf bnxt = bcur;
                if (kc + 1 < KCHUNKS) bnxt = *(const v16bf*)(bl + (kc + 1) * 512);
                c = __builtin_amdgcn_wmma_f32_16x16x32_bf16(
                        false, afrag[kc], false, bcur, (short)0, c, false, false);
                bcur = bnxt;
            }

            float e2v  = lds_e2[nt * 16 + m];
            int   myid = nt * 16 + m;
            #pragma unroll
            for (int v = 0; v < 8; ++v) {
                float sc = 2.0f * c[v] - e2v;       // argmax(2 x.e - |e|^2)
                if (sc > best[v]) { best[v] = sc; bidx[v] = myid; }
            }

            // in-order async completion: <=4 outstanding => tile nt+1 has landed
            wait_async_le<4>();
            __syncthreads();
        }

        // ---- argmax across the 16 lanes of each half ----
        #pragma unroll
        for (int off = 1; off < 16; off <<= 1) {
            #pragma unroll
            for (int v = 0; v < 8; ++v) {
                float ob = __shfl_xor(best[v], off);
                int   oi = __shfl_xor(bidx[v], off);
                if (ob > best[v] || (ob == best[v] && oi < bidx[v])) {
                    best[v] = ob; bidx[v] = oi;
                }
            }
        }
        if (m == 0) {
            #pragma unroll
            for (int v = 0; v < 8; ++v) {
                int row = r0 + half * 8 + v;
                lds_idx[row] = bidx[v];
                codes_out[(size_t)s * NTOK + (size_t)bb * TT + t0 + row] = bidx[v];
            }
        }

        // ---- gather fp32 codebook rows; residual = -(q - residual); loss += diff^2 ----
        for (int rr = 0; rr < 16; ++rr) {
            int row = r0 + rr;
            int bin = lds_idx[row];
            const float* eptr = embed + ((size_t)s * BINS + bin) * DD;
            #pragma unroll
            for (int j = 0; j < 8; ++j) {
                int d = lane + j * 32;
                float q    = eptr[d];
                float r    = (float)lds_res[row * RSTRIDE + d];
                float diff = q - r;
                lossAcc   += diff * diff;
                lds_res[row * RSTRIDE + d] = (bf16)(-diff);
            }
        }
    }

    // ---- quantized = x - final_residual ----
    __syncthreads();
    for (int i = tid; i < ROWS * DD; i += 256) {
        int row = i & (ROWS - 1);
        int d   = i >> 7;
        size_t xi = ((size_t)bb * DD + d) * TT + t0 + row;
        q_out[xi] = x[xi] - (float)lds_res[row * RSTRIDE + d];
    }

    // ---- commitment-loss partial: wave reduce, one atomic per wave ----
    #pragma unroll
    for (int off = 16; off > 0; off >>= 1) lossAcc += __shfl_down(lossAcc, off);
    if (lane == 0) atomicAdd(ws_loss, lossAcc);
}

// ---------- finalize: penalty = sum / (NQ * N * D) ----------
__global__ void rvq_fin(const float* __restrict__ ws_loss, float* __restrict__ pen) {
    *pen = ws_loss[0] * (1.0f / ((float)NQ * (float)NTOK * (float)DD));
}

extern "C" void kernel_launch(void* const* d_in, const int* in_sizes, int n_in,
                              void* d_out, int out_size, void* d_ws, size_t ws_size,
                              hipStream_t stream) {
    (void)in_sizes; (void)n_in; (void)out_size; (void)ws_size;
    const float* x     = (const float*)d_in[0];   // [B, D, T] fp32
    const float* embed = (const float*)d_in[1];   // [NQ, BINS, D] fp32

    char*  ws      = (char*)d_ws;
    float* ws_loss = (float*)ws;
    float* ws_e2   = (float*)(ws + WS_E2_OFF);
    bf16*  ws_b    = (bf16*)(ws + WS_B_OFF);

    float* q_out     = (float*)d_out;
    int*   codes_out = (int*)((float*)d_out + QUANT_ELEMS);
    float* pen_out   = (float*)d_out + QUANT_ELEMS + CODES_ELEMS;

    rvq_prep_e2 <<<NQ * BINS / 256, 256, 0, stream>>>(embed, ws_e2, ws_loss);
    rvq_prep_swz<<<NQ * BINS * DD / 256, 256, 0, stream>>>(embed, ws_b);
    rvq_main    <<<NTOK / ROWS, 256, 0, stream>>>(x, embed, ws_b, ws_e2, ws_loss,
                                                  q_out, codes_out);
    rvq_fin     <<<1, 1, 0, stream>>>(ws_loss, pen_out);
}
